// GRU_layer_51926154609337
// MI455X (gfx1250) — matrix operand order repaired and checked by reading, not compiled, not run on hardware
//
#include <hip/hip_runtime.h>
#include <math.h>

// ---- CDNA5 WMMA types -------------------------------------------------------
typedef __attribute__((ext_vector_type(16))) __bf16        v16bf;
typedef __attribute__((ext_vector_type(8)))  float         v8f;
typedef __attribute__((ext_vector_type(4)))  unsigned int  u32x4;
typedef int v4i __attribute__((vector_size(16)));          // matches builtin param type

#define GLOBAL_AS __attribute__((address_space(1)))
#define LDS_AS    __attribute__((address_space(3)))

#define B_  64
#define S_  512
#define D_  1024
#define H_  1024
#define NBLK_SCAN 16u

#if __has_builtin(__builtin_amdgcn_global_load_async_to_lds_b128)
#define ASYNC_LDS 1
#endif

__device__ __forceinline__ unsigned short f2bf(float f) {
  union { float f; unsigned int u; } x; x.f = f;
  unsigned int r = x.u + 0x7fffu + ((x.u >> 16) & 1u);   // round-to-nearest-even
  return (unsigned short)(r >> 16);
}
__device__ __forceinline__ float sigm(float x) { return 1.0f / (1.0f + expf(-x)); }

// ---- memory -> LDS staging (async DMA path on CDNA5, fallback = VGPR bounce) -
__device__ __forceinline__ void stage16(u32x4* l, const u32x4* g) {
#ifdef ASYNC_LDS
  __builtin_amdgcn_global_load_async_to_lds_b128(
      (GLOBAL_AS v4i*)(GLOBAL_AS void*)(u32x4*)g,
      (LDS_AS    v4i*)(LDS_AS    void*)l, 0, 0);
#else
  *l = *g;
#endif
}
// wait until at most N async loads remain in flight (N = loads of the *next* tile)
#ifdef ASYNC_LDS
#define ASYNC_WAIT(n) __builtin_amdgcn_s_wait_asynccnt(n)
#else
#define ASYNC_WAIT(n) ((void)0)
#endif

// ---- device-scope epoch barrier for the persistent scan kernel --------------
__device__ __forceinline__ void grid_barrier(unsigned* bar, unsigned epoch) {
  __threadfence();                 // flush this thread's stores to device scope
  __syncthreads();                 // whole block flushed
  if (threadIdx.x == 0) {
    __hip_atomic_fetch_add(bar, 1u, __ATOMIC_RELEASE, __HIP_MEMORY_SCOPE_AGENT);
    while (__hip_atomic_load(bar, __ATOMIC_ACQUIRE, __HIP_MEMORY_SCOPE_AGENT) <
           epoch * NBLK_SCAN)
      __builtin_amdgcn_s_sleep(2);
  }
  __syncthreads();
}

// ---- convert x: fp32 -> bf16 ------------------------------------------------
__global__ void cvt_x_kernel(const float* __restrict__ x, unsigned short* __restrict__ xb, int n) {
  int i = blockIdx.x * blockDim.x + threadIdx.x;
  if (i < n) xb[i] = f2bf(x[i]);
}

// ---- convert + transpose W (1024x1024): out[n][k] = bf16(in[k][n]) ----------
// grid (32, 32, 6), block (32, 8). z order: Wxr, Wxz, Wxh, Whr, Whz, Whh
__global__ void cvt_wT_kernel(const float* __restrict__ wxr, const float* __restrict__ wxz,
                              const float* __restrict__ wxh, const float* __restrict__ whr,
                              const float* __restrict__ whz, const float* __restrict__ whh,
                              unsigned short* __restrict__ wt) {
  __shared__ float tile[32][33];
  const float* src;
  switch (blockIdx.z) {
    case 0: src = wxr; break; case 1: src = wxz; break; case 2: src = wxh; break;
    case 3: src = whr; break; case 4: src = whz; break; default: src = whh; break;
  }
  unsigned short* dst = wt + (size_t)blockIdx.z * H_ * H_;
  int c = blockIdx.x * 32 + threadIdx.x;
  int r = blockIdx.y * 32 + threadIdx.y;
  #pragma unroll
  for (int i = 0; i < 32; i += 8)
    tile[threadIdx.y + i][threadIdx.x] = src[(size_t)(r + i) * H_ + c];
  __syncthreads();
  int kx = blockIdx.y * 32 + threadIdx.x;
  int ny = blockIdx.x * 32 + threadIdx.y;
  #pragma unroll
  for (int i = 0; i < 32; i += 8)
    dst[(size_t)(ny + i) * H_ + kx] = f2bf(tile[threadIdx.x][threadIdx.y + i]);
}

// ---- zero initial hidden state + barrier counter ----------------------------
__global__ void init_h_kernel(float* __restrict__ hbuf, unsigned short* __restrict__ hb,
                              unsigned* __restrict__ bar, int n) {
  int i = blockIdx.x * blockDim.x + threadIdx.x;
  if (i < n) { hbuf[i] = 0.0f; hb[i] = 0; }
  if (i == 0) *bar = 0u;
}

// ---- Phase 1: X projections, [32768 x 1024] @ [1024 x 1024] + bias ----------
// grid (N/128, M/128, 3), block 256 (8 waves). Wave w owns M-strip w (16 rows) x 128 cols.
// Double-buffered LDS: async DMA for tile k+1 overlaps the 8 WMMAs of tile k.
__global__ __launch_bounds__(256) void xproj_gemm_kernel(
    const unsigned short* __restrict__ xb,     // [M,K] bf16
    const unsigned short* __restrict__ wt,     // [3][N][K] bf16 (transposed)
    const float* __restrict__ br, const float* __restrict__ bz, const float* __restrict__ bh,
    float* __restrict__ xr, float* __restrict__ xz, float* __restrict__ outh) {
  const int K = D_;
  const int NK = K / 32;
  const int g  = blockIdx.z;
  const unsigned short* W = wt + (size_t)g * H_ * K;
  const float* bias = (g == 0) ? br : (g == 1) ? bz : bh;
  float* out        = (g == 0) ? xr : (g == 1) ? xz : outh;

  const int n0 = blockIdx.x * 128;
  const int m0 = blockIdx.y * 128;
  const int tid  = threadIdx.x;
  const int wv   = tid >> 5;
  const int lane = tid & 31;
  const int lr   = lane & 15;
  const int kh   = lane >> 4;

  __shared__ u32x4 At[2][128 * 4];   // ping-pong: 128 rows x 32 bf16
  __shared__ u32x4 Bt[2][128 * 4];

  v8f acc[8];
  #pragma unroll
  for (int j = 0; j < 8; ++j) {
    float b = bias[n0 + j * 16 + lr];
    #pragma unroll
    for (int e = 0; e < 8; ++e) acc[j][e] = b;
  }

  const int row0 = tid >> 2, row1 = (tid + 256) >> 2, seg = tid & 3;
  const u32x4* gA0 = (const u32x4*)(xb + (size_t)(m0 + row0) * K) + seg;
  const u32x4* gA1 = (const u32x4*)(xb + (size_t)(m0 + row1) * K) + seg;
  const u32x4* gB0 = (const u32x4*)(W  + (size_t)(n0 + row0) * K) + seg;
  const u32x4* gB1 = (const u32x4*)(W  + (size_t)(n0 + row1) * K) + seg;
  // k0 advance in u32x4 units: 32 bf16 = 4 x u32x4 per row
  #define XQ(p, k0) ((p) + (size_t)(k0) * 4 / 32 * 32)   // (k0/32)*4... keep explicit below

  // prologue: tile 0
  stage16(&At[0][tid],       gA0);
  stage16(&At[0][tid + 256], gA1);
  stage16(&Bt[0][tid],       gB0);
  stage16(&Bt[0][tid + 256], gB1);

  for (int it = 0; it < NK; ++it) {
    const int buf = it & 1;
    if (it + 1 < NK) {
      const int q = (it + 1) * 4;          // u32x4 offset of next K-tile within a row
      stage16(&At[buf ^ 1][tid],       gA0 + q);
      stage16(&At[buf ^ 1][tid + 256], gA1 + q);
      stage16(&Bt[buf ^ 1][tid],       gB0 + q);
      stage16(&Bt[buf ^ 1][tid + 256], gB1 + q);
      __builtin_prefetch((const void*)(gA0 + q + 4), 0, 1);   // +2 tiles ahead into L2
      __builtin_prefetch((const void*)(gB0 + q + 4), 0, 1);
      ASYNC_WAIT(4);                       // older (current-buffer) loads done
    } else {
      ASYNC_WAIT(0);
    }
    __syncthreads();
    union { v16bf v; u32x4 u[2]; } a, b;
    a.u[0] = At[buf][(wv * 16 + lr) * 4 + kh];        // K = kh*8 + 0..7
    a.u[1] = At[buf][(wv * 16 + lr) * 4 + 2 + kh];    // K = 16 + kh*8 + 0..7
    #pragma unroll
    for (int j = 0; j < 8; ++j) {
      b.u[0] = Bt[buf][(j * 16 + lr) * 4 + kh];           // K = kh*8 + 0..7
      b.u[1] = Bt[buf][(j * 16 + lr) * 4 + 2 + kh];       // K = 16 + kh*8 + 0..7
      acc[j] = __builtin_amdgcn_wmma_f32_16x16x32_bf16(
          false, a.v, false, b.v, (short)0, acc[j], false, false);
    }
    // no trailing barrier: the next iteration stages tile it+2 into this buffer
    // before its barrier, so no barrier orders those writes after other
    // threads' reads here.
  }
  #pragma unroll
  for (int j = 0; j < 8; ++j) {
    int col = n0 + j * 16 + lr;
    #pragma unroll
    for (int r = 0; r < 8; ++r) {
      int row = m0 + wv * 16 + r + 8 * kh;
      __builtin_nontemporal_store(acc[j][r], &out[(size_t)row * H_ + col]);
    }
  }
}

// ---- Phase 2: persistent scan kernel, ONE launch for all 512 steps ----------
// 16 blocks x 256 threads. Per step:
//   phase A (16 blocks = 2 gates x 8 N-tiles of 128): R/Z gates -> rh (bf16), zbuf
//   grid barrier
//   phase B (16 blocks = 16 N-tiles of 64): T = tanh(xh + RH@Whh); H = Z*H+(1-Z)*T
//   grid barrier
// Wh weights (6 MB bf16 total) stay resident in the 192 MB L2 across all steps.
__global__ __launch_bounds__(256) void gru_scan_kernel(
    const unsigned short* __restrict__ whT,    // [WhrT; WhzT; WhhT] bf16
    const float* __restrict__ xr, const float* __restrict__ xz,
    float* __restrict__ out,                   // xh in, outputs out (in-place)
    float* __restrict__ hbuf, unsigned short* __restrict__ hb,
    unsigned short* __restrict__ rh, float* __restrict__ zbuf,
    unsigned* __restrict__ bar) {
  const int K = H_;
  const int NK = K / 32;
  const int p = blockIdx.x;
  const int tid = threadIdx.x, wv = tid >> 5, lane = tid & 31, lr = lane & 15, kh = lane >> 4;

  __shared__ u32x4 Ht[2][64 * 4];     // ping-pong A tile (Hprev or RH), 64 x 32 bf16
  __shared__ u32x4 Wt[2][128 * 4];    // ping-pong W^T tile

  // phase A geometry: gate + 128-wide N tile
  const int gA  = p >> 3;
  const int n0A = (p & 7) * 128;
  const unsigned short* WA = whT + (size_t)gA * (H_ * K);
  const float* xpre = (gA == 0) ? xr : xz;
  const int msA = (wv & 3) * 16;
  const int nsA = (wv >> 2) * 64;
  // phase B geometry: 64-wide N tile
  const int n0B = p * 64;
  const unsigned short* WB = whT + (size_t)2 * (H_ * K);
  const int msB = (wv & 3) * 16;
  const int nsB = (wv >> 2) * 32;

  const int row0 = tid >> 2, row1 = (tid + 256) >> 2, seg = tid & 3;

  const u32x4* gH  = (const u32x4*)(hb + (size_t)row0 * K) + seg;            // phase A src
  const u32x4* gW0 = (const u32x4*)(WA + (size_t)(n0A + row0) * K) + seg;
  const u32x4* gW1 = (const u32x4*)(WA + (size_t)(n0A + row1) * K) + seg;
  const u32x4* gR  = (const u32x4*)(rh + (size_t)row0 * K) + seg;            // phase B src
  const u32x4* gW2 = (const u32x4*)(WB + (size_t)(n0B + row0) * K) + seg;

  unsigned epoch = 0;

  for (int t = 0; t < S_; ++t) {
    // ================= phase A: R and Z gates =================
    {
      v8f acc[4];
      #pragma unroll
      for (int j = 0; j < 4; ++j)
        #pragma unroll
        for (int e = 0; e < 8; ++e) acc[j][e] = 0.0f;

      stage16(&Ht[0][tid],       gH);
      stage16(&Wt[0][tid],       gW0);
      stage16(&Wt[0][tid + 256], gW1);
      for (int it = 0; it < NK; ++it) {
        const int buf = it & 1;
        if (it + 1 < NK) {
          const int q = (it + 1) * 4;
          stage16(&Ht[buf ^ 1][tid],       gH  + q);
          stage16(&Wt[buf ^ 1][tid],       gW0 + q);
          stage16(&Wt[buf ^ 1][tid + 256], gW1 + q);
          ASYNC_WAIT(3);
        } else {
          ASYNC_WAIT(0);
        }
        __syncthreads();
        union { v16bf v; u32x4 u[2]; } a, b;
        a.u[0] = Ht[buf][(msA + lr) * 4 + kh];
        a.u[1] = Ht[buf][(msA + lr) * 4 + 2 + kh];
        #pragma unroll
        for (int j = 0; j < 4; ++j) {
          int c = nsA + j * 16;
          b.u[0] = Wt[buf][(c + lr) * 4 + kh];
          b.u[1] = Wt[buf][(c + lr) * 4 + 2 + kh];
          acc[j] = __builtin_amdgcn_wmma_f32_16x16x32_bf16(
              false, a.v, false, b.v, (short)0, acc[j], false, false);
        }
      }
      #pragma unroll
      for (int j = 0; j < 4; ++j) {
        int col = n0A + nsA + j * 16 + lr;
        #pragma unroll
        for (int r = 0; r < 8; ++r) {
          int row = msA + r + 8 * kh;                        // batch index
          size_t gidx = ((size_t)row * S_ + t) * H_ + col;
          float sv = sigm(acc[j][r] + xpre[gidx]);
          size_t hidx = (size_t)row * H_ + col;
          if (gA == 0) rh[hidx] = f2bf(sv * hbuf[hidx]);
          else         zbuf[hidx] = sv;
        }
      }
    }
    grid_barrier(bar, ++epoch);
    // ================= phase B: candidate + blend =================
    {
      v8f acc[2];
      #pragma unroll
      for (int j = 0; j < 2; ++j)
        #pragma unroll
        for (int e = 0; e < 8; ++e) acc[j][e] = 0.0f;

      stage16(&Ht[0][tid], gR);
      stage16(&Wt[0][tid], gW2);
      for (int it = 0; it < NK; ++it) {
        const int buf = it & 1;
        if (it + 1 < NK) {
          const int q = (it + 1) * 4;
          stage16(&Ht[buf ^ 1][tid], gR  + q);
          stage16(&Wt[buf ^ 1][tid], gW2 + q);
          ASYNC_WAIT(2);
        } else {
          ASYNC_WAIT(0);
        }
        __syncthreads();
        union { v16bf v; u32x4 u[2]; } a, b;
        a.u[0] = Ht[buf][(msB + lr) * 4 + kh];
        a.u[1] = Ht[buf][(msB + lr) * 4 + 2 + kh];
        #pragma unroll
        for (int j = 0; j < 2; ++j) {
          int c = nsB + j * 16;
          b.u[0] = Wt[buf][(c + lr) * 4 + kh];
          b.u[1] = Wt[buf][(c + lr) * 4 + 2 + kh];
          acc[j] = __builtin_amdgcn_wmma_f32_16x16x32_bf16(
              false, a.v, false, b.v, (short)0, acc[j], false, false);
        }
      }
      #pragma unroll
      for (int j = 0; j < 2; ++j) {
        int col = n0B + nsB + j * 16 + lr;
        #pragma unroll
        for (int r = 0; r < 8; ++r) {
          int row = msB + r + 8 * kh;
          size_t gidx = ((size_t)row * S_ + t) * H_ + col;
          float tv = tanhf(acc[j][r] + out[gidx]);           // out[gidx] holds xh_t
          size_t hidx = (size_t)row * H_ + col;
          float z  = zbuf[hidx];
          float hn = z * hbuf[hidx] + (1.0f - z) * tv;
          __builtin_nontemporal_store(hn, &out[gidx]);       // streamed, never re-read
          hbuf[hidx] = hn;
          hb[hidx]   = f2bf(hn);
        }
      }
    }
    grid_barrier(bar, ++epoch);
  }
}

// ---- final hidden state copy ------------------------------------------------
__global__ void copy_hfinal_kernel(const float* __restrict__ hbuf, float* __restrict__ dst, int n) {
  int i = blockIdx.x * blockDim.x + threadIdx.x;
  if (i < n) dst[i] = hbuf[i];
}

// ---- launcher ---------------------------------------------------------------
extern "C" void kernel_launch(void* const* d_in, const int* in_sizes, int n_in,
                              void* d_out, int out_size, void* d_ws, size_t ws_size,
                              hipStream_t stream) {
  (void)in_sizes; (void)n_in; (void)out_size; (void)ws_size;
  const float* x   = (const float*)d_in[0];
  const float* Wxr = (const float*)d_in[1];
  const float* Whr = (const float*)d_in[2];
  const float* br  = (const float*)d_in[3];
  const float* Wxz = (const float*)d_in[4];
  const float* Whz = (const float*)d_in[5];
  const float* bz  = (const float*)d_in[6];
  const float* Wxh = (const float*)d_in[7];
  const float* Whh = (const float*)d_in[8];
  const float* bh  = (const float*)d_in[9];
  float* out = (float*)d_out;

  // workspace layout (bytes, all offsets 256-aligned)
  char* ws = (char*)d_ws;
  unsigned short* xb   = (unsigned short*)(ws);                        //  64 MiB x bf16
  unsigned short* wt   = (unsigned short*)(ws + 67108864ull);          //  12 MiB 6x W^T bf16
  float*          xr   = (float*)(ws + 79691776ull);                   // 128 MiB
  float*          xz   = (float*)(ws + 213909504ull);                  // 128 MiB
  float*          hbuf = (float*)(ws + 348127232ull);                  // 256 KiB
  unsigned short* hb   = (unsigned short*)(ws + 348389376ull);         // 128 KiB
  unsigned short* rh   = (unsigned short*)(ws + 348520448ull);         // 128 KiB
  float*          zbuf = (float*)(ws + 348651520ull);                  // 256 KiB
  unsigned*       bar  = (unsigned*)(ws + 348913664ull);               //   4 B

  const int nx = B_ * S_ * D_;
  cvt_x_kernel<<<nx / 256, 256, 0, stream>>>(x, xb, nx);
  cvt_wT_kernel<<<dim3(32, 32, 6), dim3(32, 8), 0, stream>>>(Wxr, Wxz, Wxh, Whr, Whz, Whh, wt);
  init_h_kernel<<<(B_ * H_) / 256, 256, 0, stream>>>(hbuf, hb, bar, B_ * H_);

  // input projections: gate h writes directly into d_out (consumed in-place by scan)
  xproj_gemm_kernel<<<dim3(H_ / 128, (B_ * S_) / 128, 3), 256, 0, stream>>>(
      xb, wt, br, bz, bh, xr, xz, out);

  // whole recurrent scan in a single persistent launch
  const unsigned short* whT = wt + (size_t)3 * H_ * H_;   // [WhrT; WhzT; WhhT]
  gru_scan_kernel<<<dim3(NBLK_SCAN, 1, 1), 256, 0, stream>>>(
      whT, xr, xz, out, hbuf, hb, rh, zbuf, bar);

  copy_hfinal_kernel<<<(B_ * H_) / 256, 256, 0, stream>>>(hbuf, out + (size_t)B_ * S_ * H_, B_ * H_);
}
